// GumbelIBFilter_64381559767319
// MI455X (gfx1250) — matrix operand order repaired and checked by reading, not compiled, not run on hardware
//
#include <hip/hip_runtime.h>

typedef __attribute__((ext_vector_type(16))) __bf16 v16bf;
typedef __attribute__((ext_vector_type(8)))  __bf16 v8bf;
typedef __attribute__((ext_vector_type(8)))  float  v8f;
typedef unsigned int u32x4 __attribute__((ext_vector_type(4)));
typedef int          i32x8 __attribute__((ext_vector_type(8)));
typedef int          i32x4 __attribute__((ext_vector_type(4)));

#define SELECT_K 50
#define BETA     0.001f
#define B_       32
#define K_       8192
#define DC       256
#define DQ       256
#define INDIM    512      // DC + DQ
#define M_ENC    1600     // B_ * SELECT_K

#if __has_builtin(__builtin_amdgcn_tensor_load_to_lds) && __has_builtin(__builtin_amdgcn_s_wait_tensorcnt)
#define HAS_TDM 1
#else
#define HAS_TDM 0
#endif

// ---- workspace layout (bytes), all section bases 64B-aligned ----
#define WS_LOGITS 0u              // f32  [32][8192]    1,048,576
#define WS_W1T    1048576u        // bf16 [256][512]      262,144
#define WS_WE1T   1310720u        // bf16 [512][512]      524,288
#define WS_WE2T   1835008u        // bf16 [1024][512]   1,048,576
#define WS_TOPK   2883584u        // i32  [32][50]          6,400
#define WS_HE     2889984u        // bf16 [1600][512]   1,638,400
#define WS_PARAMS 4528384u        // f32  [1600][1024]  6,553,600

// ---- output layout (float elements, concatenated in return order) ----
#define OUT_Z    0                // [32,50,512]   819,200
#define OUT_SEL  819200           // [32,50,8192]  13,107,200  (== hard, forward)
#define OUT_KL   13926400         // scalar
#define OUT_HARD 13926401         // [32,50,8192]  13,107,200
#define OUT_SS   27033601         // [32,50]       1,600
#define OUT_SRS  27035201         // [32,50]       1,600

// ---------------------------------------------------------------------------
// A-operand handling: 16x32 bf16 tile. Lane = row (l&15); elements 0..7 hold
// K = cb + (l>>4)*16 .. +7, elements 8..15 hold K = cb+(l>>4)*16+8 .. +7.
// x-row = concat(cand_row[0:256], query_row[0:256]); runs of 8 never straddle
// the 256 boundary. Raw f32 fragments are prefetched one chunk ahead.
// ---------------------------------------------------------------------------
struct ARaw { float4 x0, x1, y0, y1; };

__device__ inline ARaw load_a_raw(const float* __restrict__ rowc,
                                  const float* __restrict__ rowq,
                                  int cb, int hi) {
  int g0 = cb + hi * 16;
  int g1 = cb + hi * 16 + 8;
  const float* p0 = (g0 < DC) ? (rowc + g0) : (rowq + (g0 - DC));
  const float* p1 = (g1 < DC) ? (rowc + g1) : (rowq + (g1 - DC));
  ARaw r;
  r.x0 = ((const float4*)p0)[0]; r.x1 = ((const float4*)p0)[1];
  r.y0 = ((const float4*)p1)[0]; r.y1 = ((const float4*)p1)[1];
  return r;
}

__device__ inline v16bf cvt_a(const ARaw& r) {
  v16bf a;
  a[0]  = (__bf16)r.x0.x; a[1]  = (__bf16)r.x0.y; a[2]  = (__bf16)r.x0.z; a[3]  = (__bf16)r.x0.w;
  a[4]  = (__bf16)r.x1.x; a[5]  = (__bf16)r.x1.y; a[6]  = (__bf16)r.x1.z; a[7]  = (__bf16)r.x1.w;
  a[8]  = (__bf16)r.y0.x; a[9]  = (__bf16)r.y0.y; a[10] = (__bf16)r.y0.z; a[11] = (__bf16)r.y0.w;
  a[12] = (__bf16)r.y1.x; a[13] = (__bf16)r.y1.y; a[14] = (__bf16)r.y1.z; a[15] = (__bf16)r.y1.w;
  return a;
}

#if HAS_TDM
// TDM: stage a [256 rows x 32 bf16] tile (row stride 512 bf16) from global to
// LDS at ldsoff. D# per CDNA5 ISA: group0 = count/lds/
// global_addr/type=2; group1 = data_size=2B, tensor_dim0=512, tensor_dim1=256,
// tile_dim0=32, tile_dim1=256, tensor_dim0_stride=512; groups 2/3/4 zero (2D).
// This toolchain's builtin is the 6-arg form (g0, g1, g2, g3, g4, cpol).
__device__ inline void tdm_stage(const __bf16* gsrc, unsigned ldsoff) {
  unsigned long long ga = (unsigned long long)gsrc;
  u32x4 g0;
  g0[0] = 1u;                                               // count=1, user mode
  g0[1] = ldsoff;                                           // lds_addr
  g0[2] = (unsigned)ga;                                     // global_addr[31:0]
  g0[3] = (unsigned)((ga >> 32) & 0x01FFFFFFu) | (2u << 30);// [56:32] | type=2
  i32x8 g1;
  g1[0] = 0x00010000;        // workgroup_mask=0, data_size=1 (2 bytes)
  g1[1] = (int)(512u << 16); // atomic_barrier_addr=0 | tensor_dim0[15:0]=512
  g1[2] = (int)(256u << 16); // tensor_dim0[31:16]=0 | tensor_dim1[15:0]=256
  g1[3] = (int)(32u << 16);  // tensor_dim1[31:16]=0 | tile_dim0=32
  g1[4] = 256;               // tile_dim1=256 | tile_dim2=0
  g1[5] = 512;               // tensor_dim0_stride[31:0]=512
  g1[6] = 0;                 // stride0 hi | tensor_dim1_stride lo
  g1[7] = 0;
  i32x4 zz4 = {0, 0, 0, 0};
  i32x8 zz8 = {0, 0, 0, 0, 0, 0, 0, 0};
  __builtin_amdgcn_tensor_load_to_lds(g0, g1, zz4, zz4, zz8, 0);
}
#endif

// ---- K0: transpose weights to bf16 (N-major, K-contiguous) + zero KL slot ----
__global__ __launch_bounds__(256) void k_prep(
    const float* __restrict__ W1, const float* __restrict__ We1,
    const float* __restrict__ We2, __bf16* __restrict__ w1t,
    __bf16* __restrict__ we1t, __bf16* __restrict__ we2t,
    float* __restrict__ out) {
  int t = blockIdx.x * 256 + threadIdx.x;
  if (t == 0) out[OUT_KL] = 0.f;
  int n = t >> 9, k = t & 511;
  if (t < 256 * 512)  w1t[t]  = (__bf16)W1[k * 256 + n];
  if (t < 512 * 512)  we1t[t] = (__bf16)We1[k * 512 + n];
  if (t < 1024 * 512) we2t[t] = (__bf16)We2[(size_t)k * 1024 + n];
}

// ---- K1: scorer MLP via bf16 WMMA; logits[b][k] -> ws ----
// grid = 32*64 blocks, 256 thr (8 waves); block = 128 rows, wave = 16x256.
// Double-buffered LDS W1^T chunks (TDM-staged when available), A prefetch one
// chunk ahead, B tiles batch-preloaded so WMMAs issue back-to-back.
__global__ __launch_bounds__(256) void k_scorer(
    const float* __restrict__ query, const float* __restrict__ cand,
    const float* __restrict__ b1, const float* __restrict__ W2,
    const float* __restrict__ b2, const __bf16* __restrict__ w1t,
    float* __restrict__ logits) {
  __shared__ __bf16 w1s[2][256 * 32];         // 2 x 16KB double buffer
  const int tid = threadIdx.x;
  const int lane = tid & 31, wv = tid >> 5;
  const int hi = lane >> 4, lr = lane & 15;
  const int b  = blockIdx.x >> 6;
  const int k0 = (blockIdx.x & 63) * 128 + wv * 16;
  const float* rowc = cand + ((size_t)b * K_ + k0 + lr) * DC;
  const float* rowq = query + (size_t)b * DQ;

  v8f acc[16];
  v8f zero = {0.f, 0.f, 0.f, 0.f, 0.f, 0.f, 0.f, 0.f};
#pragma unroll
  for (int nt = 0; nt < 16; ++nt) acc[nt] = zero;

#if HAS_TDM
  const unsigned lds_off[2] = {(unsigned)(unsigned long long)&w1s[0][0],
                               (unsigned)(unsigned long long)&w1s[1][0]};
  if (wv == 0) tdm_stage(w1t, lds_off[0]);     // stage chunk 0
#else
  {
    const uint4* src = (const uint4*)(w1t + (size_t)tid * INDIM);
    uint4* dst = (uint4*)(&w1s[0][tid * 32]);
    dst[0] = src[0]; dst[1] = src[1];
  }
#endif
  ARaw ar = load_a_raw(rowc, rowq, 0, hi);     // prefetch A chunk 0

#pragma unroll 1
  for (int c = 0; c < 16; ++c) {
    const int cb = c * 32;
#if HAS_TDM
    if (wv == 0) __builtin_amdgcn_s_wait_tensorcnt(0);
#endif
    __syncthreads();  // buf[c&1] staged; buf[1-(c&1)] no longer being read
    if (c < 15) {     // stage next chunk while this chunk computes
#if HAS_TDM
      if (wv == 0) tdm_stage(w1t + (cb + 32), lds_off[1 - (c & 1)]);
#else
      const uint4* src = (const uint4*)(w1t + (size_t)tid * INDIM + cb + 32);
      uint4* dst = (uint4*)(&w1s[1 - (c & 1)][tid * 32]);
      dst[0] = src[0]; dst[1] = src[1];
#endif
    }
    v16bf av = cvt_a(ar);
    if (c < 15) ar = load_a_raw(rowc, rowq, cb + 32, hi);  // prefetch next A

    const __bf16* wb = &w1s[c & 1][0];
    v16bf bvs[16];
#pragma unroll
    for (int nt = 0; nt < 16; ++nt)
      bvs[nt] = *(const v16bf*)(wb + (nt * 16 + lr) * 32 + hi * 16);
#pragma unroll
    for (int nt = 0; nt < 16; ++nt)
      acc[nt] = __builtin_amdgcn_wmma_f32_16x16x32_bf16(
          false, av, false, bvs[nt], (short)0, acc[nt], false, false);
  }

  // epilogue: logits = relu(h + b1) . W2 + b2, reduced across the 16-lane group
  float plog[8];
#pragma unroll
  for (int i = 0; i < 8; ++i) plog[i] = 0.f;
#pragma unroll
  for (int nt = 0; nt < 16; ++nt) {
    int n = nt * 16 + lr;
    float bb = b1[n], ww = W2[n];
#pragma unroll
    for (int i = 0; i < 8; ++i) {
      float h = acc[nt][i] + bb;
      h = h > 0.f ? h : 0.f;
      plog[i] += h * ww;
    }
  }
  float b2v = b2[0];
#pragma unroll
  for (int i = 0; i < 8; ++i) {
    float v = plog[i];
    v += __shfl_xor(v, 1, 32);
    v += __shfl_xor(v, 2, 32);
    v += __shfl_xor(v, 4, 32);
    v += __shfl_xor(v, 8, 32);
    if (lr == 0) logits[(size_t)b * K_ + k0 + i + hi * 8] = v + b2v;
  }
}

// ---- K2: gumbel top-50 per batch row; also gather selected scores ----
__global__ __launch_bounds__(256) void k_topk(
    const float* __restrict__ logits, const float* __restrict__ gumbel,
    const float* __restrict__ cand_score, int* __restrict__ topk,
    float* __restrict__ out) {
  __shared__ float vals[K_];
  __shared__ float rmax[256];
  __shared__ int   ridx[256];
  const int b = blockIdx.x, t = threadIdx.x;
  for (int i = t; i < K_; i += 256)
    vals[i] = logits[(size_t)b * K_ + i] + gumbel[(size_t)b * K_ + i];
  __syncthreads();
  for (int it = 0; it < SELECT_K; ++it) {
    float m = -3.0e38f; int mi = 0;
    for (int i = t; i < K_; i += 256) {
      float v = vals[i];
      if (v > m) { m = v; mi = i; }
    }
    rmax[t] = m; ridx[t] = mi;
    __syncthreads();
    for (int s = 128; s > 0; s >>= 1) {
      if (t < s) {
        float mo = rmax[t + s]; int io = ridx[t + s];
        if (mo > rmax[t] || (mo == rmax[t] && io < ridx[t])) { rmax[t] = mo; ridx[t] = io; }
      }
      __syncthreads();
    }
    if (t == 0) {
      int gi = ridx[0];
      topk[b * SELECT_K + it] = gi;
      out[OUT_SS  + b * SELECT_K + it] = logits[(size_t)b * K_ + gi];
      out[OUT_SRS + b * SELECT_K + it] = cand_score[(size_t)b * K_ + gi];
      vals[gi] = -3.0e38f;
    }
    __syncthreads();
  }
}

// ---- K3: one-hot outputs (selection == hard in forward) ----
__global__ __launch_bounds__(256) void k_onehot(
    const int* __restrict__ topk, float* __restrict__ out) {
  size_t t = (size_t)blockIdx.x * 256 + threadIdx.x;  // [0, 32*50*8192)
  int bj = (int)(t >> 13);
  int n  = (int)(t & 8191);
  float v = (n == topk[bj]) ? 1.0f : 0.0f;
  out[OUT_SEL  + t] = v;
  out[OUT_HARD + t] = v;
}

// ---- K4: encoder layer 1: he = relu([sel||q] @ We1 + be1) -> bf16 ws ----
__global__ __launch_bounds__(256) void k_enc1(
    const float* __restrict__ query, const float* __restrict__ cand,
    const float* __restrict__ be1, const __bf16* __restrict__ we1t,
    const int* __restrict__ topk, __bf16* __restrict__ he) {
  const int tid = threadIdx.x;
  const int lane = tid & 31, wv = tid >> 5;
  const int hi = lane >> 4, lr = lane & 15;
  const int r0 = blockIdx.x * 16;
  const int r = r0 + lr;
  const int b = r / SELECT_K, j = r % SELECT_K;
  const int kidx = topk[b * SELECT_K + j];
  const float* rowc = cand + ((size_t)b * K_ + kidx) * DC;
  const float* rowq = query + (size_t)b * DQ;

  v8f acc[4];
  v8f zero = {0.f, 0.f, 0.f, 0.f, 0.f, 0.f, 0.f, 0.f};
#pragma unroll
  for (int c = 0; c < 4; ++c) acc[c] = zero;

  ARaw ar = load_a_raw(rowc, rowq, 0, hi);
#pragma unroll 1
  for (int c = 0; c < 16; ++c) {
    const int cb = c * 32;
    v16bf av = cvt_a(ar);
    if (c < 15) ar = load_a_raw(rowc, rowq, cb + 32, hi);
    v16bf bvs[4];
#pragma unroll
    for (int q = 0; q < 4; ++q) {
      int n = (wv * 4 + q) * 16 + lr;
      bvs[q] = *(const v16bf*)(we1t + (size_t)n * INDIM + cb + hi * 16);
    }
#pragma unroll
    for (int q = 0; q < 4; ++q)
      acc[q] = __builtin_amdgcn_wmma_f32_16x16x32_bf16(
          false, av, false, bvs[q], (short)0, acc[q], false, false);
  }
#pragma unroll
  for (int q = 0; q < 4; ++q) {
    int n = (wv * 4 + q) * 16 + lr;
    float bb = be1[n];
#pragma unroll
    for (int i = 0; i < 8; ++i) {
      float h = acc[q][i] + bb;
      h = h > 0.f ? h : 0.f;
      he[(size_t)(r0 + i + hi * 8) * 512 + n] = (__bf16)h;
    }
  }
}

// ---- K5: encoder layer 2: params = he @ We2 + be2 -> f32 ws ----
__global__ __launch_bounds__(256) void k_enc2(
    const __bf16* __restrict__ he, const __bf16* __restrict__ we2t,
    const float* __restrict__ be2, float* __restrict__ params) {
  const int tid = threadIdx.x;
  const int lane = tid & 31, wv = tid >> 5;
  const int hi = lane >> 4, lr = lane & 15;
  const int r0 = blockIdx.x * 16;
  const int r = r0 + lr;

  v8f acc[8];
  v8f zero = {0.f, 0.f, 0.f, 0.f, 0.f, 0.f, 0.f, 0.f};
#pragma unroll
  for (int c = 0; c < 8; ++c) acc[c] = zero;

#pragma unroll 1
  for (int cb = 0; cb < 512; cb += 32) {
    v16bf av;
    const __bf16* p0 = he + (size_t)r * 512 + cb + hi * 16;
    v8bf lo  = *(const v8bf*)p0;
    v8bf hi8 = *(const v8bf*)(p0 + 8);
#pragma unroll
    for (int e = 0; e < 8; ++e) { av[e] = lo[e]; av[8 + e] = hi8[e]; }
    v16bf bvs[8];
#pragma unroll
    for (int q = 0; q < 8; ++q) {
      int n = (wv * 8 + q) * 16 + lr;
      bvs[q] = *(const v16bf*)(we2t + (size_t)n * 512 + cb + hi * 16);
    }
#pragma unroll
    for (int q = 0; q < 8; ++q)
      acc[q] = __builtin_amdgcn_wmma_f32_16x16x32_bf16(
          false, av, false, bvs[q], (short)0, acc[q], false, false);
  }
#pragma unroll
  for (int q = 0; q < 8; ++q) {
    int n = (wv * 8 + q) * 16 + lr;
    float bb = be2[n];
#pragma unroll
    for (int i = 0; i < 8; ++i)
      params[(size_t)(r0 + i + hi * 8) * 1024 + n] = acc[q][i] + bb;
  }
}

// ---- K6: z = mu + exp(clip(ls))*eps ; KL reduction ----
__global__ __launch_bounds__(256) void k_z_kl(
    const float* __restrict__ params, const float* __restrict__ eps,
    float* __restrict__ out) {
  __shared__ float red[256];
  size_t t = (size_t)blockIdx.x * 256 + threadIdx.x;  // [0, 1600*512)
  int r = (int)(t >> 9);
  int d = (int)(t & 511);
  float mu = params[(size_t)r * 1024 + d];
  float ls = params[(size_t)r * 1024 + 512 + d];
  float lsc = fminf(fmaxf(ls, -10.f), 10.f);
  float sd = __expf(lsc);
  out[OUT_Z + t] = mu + sd * eps[t];
  float kl = 0.5f * (mu * mu + sd * sd - 1.f - 2.f * ls);
  red[threadIdx.x] = kl;
  __syncthreads();
  for (int s = 128; s > 0; s >>= 1) {
    if (threadIdx.x < s) red[threadIdx.x] += red[threadIdx.x + s];
    __syncthreads();
  }
  if (threadIdx.x == 0)
    atomicAdd(&out[OUT_KL], red[0] * (BETA / (float)M_ENC));
}

extern "C" void kernel_launch(void* const* d_in, const int* in_sizes, int n_in,
                              void* d_out, int out_size, void* d_ws, size_t ws_size,
                              hipStream_t stream) {
  (void)in_sizes; (void)n_in; (void)out_size; (void)ws_size;
  const float* query  = (const float*)d_in[0];
  const float* cand   = (const float*)d_in[1];
  const float* cscore = (const float*)d_in[2];
  const float* gumbel = (const float*)d_in[3];
  const float* eps    = (const float*)d_in[4];
  const float* W1     = (const float*)d_in[5];
  const float* b1     = (const float*)d_in[6];
  const float* W2     = (const float*)d_in[7];
  const float* b2     = (const float*)d_in[8];
  const float* We1    = (const float*)d_in[9];
  const float* be1    = (const float*)d_in[10];
  const float* We2    = (const float*)d_in[11];
  const float* be2    = (const float*)d_in[12];
  float* out = (float*)d_out;

  char* ws = (char*)d_ws;
  float*  logits = (float*)(ws + WS_LOGITS);
  __bf16* w1t    = (__bf16*)(ws + WS_W1T);
  __bf16* we1t   = (__bf16*)(ws + WS_WE1T);
  __bf16* we2t   = (__bf16*)(ws + WS_WE2T);
  int*    topk   = (int*)(ws + WS_TOPK);
  __bf16* he     = (__bf16*)(ws + WS_HE);
  float*  params = (float*)(ws + WS_PARAMS);

  k_prep<<<2048, 256, 0, stream>>>(W1, We1, We2, w1t, we1t, we2t, out);
  k_scorer<<<B_ * (K_ / 128), 256, 0, stream>>>(query, cand, b1, W2, b2, w1t, logits);
  k_topk<<<B_, 256, 0, stream>>>(logits, gumbel, cscore, topk, out);
  k_onehot<<<(B_ * SELECT_K * K_) / 256, 256, 0, stream>>>(topk, out);
  k_enc1<<<M_ENC / 16, 256, 0, stream>>>(query, cand, be1, we1t, topk, he);
  k_enc2<<<M_ENC / 16, 256, 0, stream>>>(he, we2t, be2, params);
  k_z_kl<<<(M_ENC * 512) / 256, 256, 0, stream>>>(params, eps, out);
}
